// QuantizedLinear_5085241279143
// MI455X (gfx1250) — compile-verified, matrix operand order checked by reading, not run on hardware
//
#include <hip/hip_runtime.h>
#include <hip/hip_bf16.h>

typedef _Float16 v16h  __attribute__((ext_vector_type(16)));
typedef _Float16 v8h_t __attribute__((ext_vector_type(8)));
typedef float    v8f   __attribute__((ext_vector_type(8)));

#define TILE_M 128
#define TILE_N 256
#define KSTEP  64
#define LDSTR  72   // padded LDS row stride in halves (144B, 16B-aligned)

#define GK 4096
#define GN 11008
#define GM 4096

__global__ __launch_bounds__(256)
void QuantizedLinear_5085241279143_kernel(const float* __restrict__ x,
                                          const int*   __restrict__ w,
                                          const float* __restrict__ scale,
                                          float*       __restrict__ out)
{
    __shared__ _Float16 sA[TILE_M * LDSTR];   // 18.4 KB
    __shared__ _Float16 sB[TILE_N * LDSTR];   // 36.9 KB

    const int nBase = blockIdx.x * TILE_N;
    const int mBase = blockIdx.y * TILE_M;

    const int t    = threadIdx.x;
    // Loader mapping: each thread owns 8 contiguous K-halves -> b128 LDS stores.
    const int colg = (t & 7) * 8;    // half/float column group within 64-wide stage
    const int row0 = t >> 3;         // 0..31, +32 per pass

    const int wave  = t >> 5;              // 8 waves (wave32)
    const int waveM = (wave >> 2) * 64;    // 2 waves along M  (128)
    const int waveN = (wave & 3)  * 64;    // 4 waves along N  (256)
    const int lane  = t & 31;
    const int l16   = lane & 15;
    const int half8 = (lane >> 4) * 8;     // K sub-chunk select per ISA A/B layout

    v8f acc[4][4];
#pragma unroll
    for (int i = 0; i < 4; ++i)
#pragma unroll
        for (int j = 0; j < 4; ++j)
            acc[i][j] = v8f{0.f, 0.f, 0.f, 0.f, 0.f, 0.f, 0.f, 0.f};

    const float* xp = x + (size_t)mBase * GK;
    const int*   wp = w + (size_t)nBase * GK;

    // ---- prologue: prefetch stage 0 into registers ----
    float4 av[4][2];   // A: 128 rows / 32 per pass = 4 passes, 8 floats each
    int4   bv[8][2];   // B: 256 rows / 32 per pass = 8 passes, 8 ints each
#pragma unroll
    for (int p = 0; p < 4; ++p) {
        const float* g = xp + (size_t)(row0 + p * 32) * GK + colg;
        av[p][0] = *(const float4*)g;
        av[p][1] = *(const float4*)(g + 4);
    }
#pragma unroll
    for (int p = 0; p < 8; ++p) {
        const int* g = wp + (size_t)(row0 + p * 32) * GK + colg;
        bv[p][0] = *(const int4*)g;
        bv[p][1] = *(const int4*)(g + 4);
    }

    for (int kb = 0; kb < GK; kb += KSTEP) {
        __syncthreads();   // previous stage's compute done reading LDS
        // ---- convert staged registers -> LDS (f16), 16B ds_store_b128 ----
#pragma unroll
        for (int p = 0; p < 4; ++p) {
            v8h_t ah = { (_Float16)av[p][0].x, (_Float16)av[p][0].y,
                         (_Float16)av[p][0].z, (_Float16)av[p][0].w,
                         (_Float16)av[p][1].x, (_Float16)av[p][1].y,
                         (_Float16)av[p][1].z, (_Float16)av[p][1].w };
            *(v8h_t*)(&sA[(row0 + p * 32) * LDSTR + colg]) = ah;
        }
#pragma unroll
        for (int p = 0; p < 8; ++p) {
            v8h_t bh = { (_Float16)(float)bv[p][0].x, (_Float16)(float)bv[p][0].y,
                         (_Float16)(float)bv[p][0].z, (_Float16)(float)bv[p][0].w,
                         (_Float16)(float)bv[p][1].x, (_Float16)(float)bv[p][1].y,
                         (_Float16)(float)bv[p][1].z, (_Float16)(float)bv[p][1].w };
            *(v8h_t*)(&sB[(row0 + p * 32) * LDSTR + colg]) = bh;
        }
        __syncthreads();

        // ---- prefetch NEXT stage while this stage computes ----
        const int kn = kb + KSTEP;
        if (kn < GK) {
#pragma unroll
            for (int p = 0; p < 4; ++p) {
                const float* g = xp + (size_t)(row0 + p * 32) * GK + kn + colg;
                av[p][0] = *(const float4*)g;
                av[p][1] = *(const float4*)(g + 4);
            }
#pragma unroll
            for (int p = 0; p < 8; ++p) {
                const int* g = wp + (size_t)(row0 + p * 32) * GK + kn + colg;
                bv[p][0] = *(const int4*)g;
                bv[p][1] = *(const int4*)(g + 4);
            }
        }

        // ---- compute: 2 K-chunks of 32, 4x4 tiles of 16x16 per wave ----
#pragma unroll
        for (int kc = 0; kc < 2; ++kc) {
            const int k0 = kc * 32 + half8;
            v16h afr[4], bfr[4];
#pragma unroll
            for (int i = 0; i < 4; ++i) {
                const _Float16* p0 = &sA[(waveM + i * 16 + l16) * LDSTR + k0];
                v8h_t lo = *(const v8h_t*)p0;        // K = k0 .. k0+7
                v8h_t hi = *(const v8h_t*)(p0 + 16); // K = k0+16 .. k0+23
                afr[i] = __builtin_shufflevector(lo, hi,
                         0,1,2,3,4,5,6,7,8,9,10,11,12,13,14,15);
            }
#pragma unroll
            for (int j = 0; j < 4; ++j) {
                const _Float16* p0 = &sB[(waveN + j * 16 + l16) * LDSTR + k0];
                v8h_t lo = *(const v8h_t*)p0;
                v8h_t hi = *(const v8h_t*)(p0 + 16);
                bfr[j] = __builtin_shufflevector(lo, hi,
                         0,1,2,3,4,5,6,7,8,9,10,11,12,13,14,15);
            }
#pragma unroll
            for (int i = 0; i < 4; ++i)
#pragma unroll
                for (int j = 0; j < 4; ++j)
                    acc[i][j] = __builtin_amdgcn_wmma_f32_16x16x32_f16(
                        /*neg_a=*/false, afr[i],
                        /*neg_b=*/false, bfr[j],
                        /*c_mod=*/(short)0, acc[i][j],
                        /*reuse_a=*/false, /*reuse_b=*/false);
        }
    }

    // ---- epilogue: per-output-channel scale, non-temporal stores ----
#pragma unroll
    for (int j = 0; j < 4; ++j) {
        const int n = nBase + waveN + j * 16 + l16;  // C/D: N = lane&15
        const float s = scale[n];
#pragma unroll
        for (int i = 0; i < 4; ++i) {
            const int mrow = mBase + waveM + i * 16 + (lane >> 4) * 8; // C/D: M = r + 8*(lane>=16)
#pragma unroll
            for (int r = 0; r < 8; ++r) {
                __builtin_nontemporal_store(acc[i][j][r] * s,
                                            &out[(size_t)(mrow + r) * GN + n]);
            }
        }
    }
}

extern "C" void kernel_launch(void* const* d_in, const int* in_sizes, int n_in,
                              void* d_out, int out_size, void* d_ws, size_t ws_size,
                              hipStream_t stream) {
    const float* x  = (const float*)d_in[0];   // [8,512,4096] f32
    const int*   w  = (const int*)d_in[1];     // [11008,4096] int32 (int8 values)
    const float* sc = (const float*)d_in[2];   // [11008] f32
    float* out = (float*)d_out;                // [8,512,11008] f32

    dim3 grid(GN / TILE_N, GM / TILE_M);       // (43, 32)
    QuantizedLinear_5085241279143_kernel<<<grid, 256, 0, stream>>>(x, w, sc, out);
}
